// Structure_Decoder_39032662786657
// MI455X (gfx1250) — matrix operand order repaired and checked by reading, not compiled, not run on hardware
//
#include <hip/hip_runtime.h>
#include <hip/hip_bf16.h>

#define NN   12288
#define EE   196608
#define NH   64

typedef __attribute__((ext_vector_type(16))) _Float16 v16h;
typedef __attribute__((ext_vector_type(8)))  _Float16 v8h;
typedef __attribute__((ext_vector_type(8)))  float    v8f;

// ---------------- stage 0: zero the accumulator region of ws ----------------
__global__ void zero_kernel(float* __restrict__ p, int n) {
    int i = blockIdx.x * blockDim.x + threadIdx.x;
    if (i < n) p[i] = 0.0f;
}

// ---------------- stage 1: degree counts (float atomics) --------------------
__global__ void degree_kernel(const int* __restrict__ src, const int* __restrict__ dst,
                              float* __restrict__ degO, float* __restrict__ degI) {
    int e = blockIdx.x * blockDim.x + threadIdx.x;
    if (e < EE) {
        atomicAdd(&degO[src[e]], 1.0f);
        atomicAdd(&degI[dst[e]], 1.0f);
    }
}

// ---------------- stage 2: deg -> clamp(deg,1)^-0.5 in place ----------------
__global__ void norm_kernel(float* __restrict__ degO, float* __restrict__ degI) {
    int i = blockIdx.x * blockDim.x + threadIdx.x;
    if (i < NN) {
        degO[i] = rsqrtf(fmaxf(degO[i], 1.0f));
        degI[i] = rsqrtf(fmaxf(degI[i], 1.0f));
    }
}

// ------- stage 3: agg[dst,f] += h[src,f]*norm_src[src]  (edge x feat) -------
__global__ void scatter_kernel(const float* __restrict__ h,
                               const int* __restrict__ src, const int* __restrict__ dst,
                               const float* __restrict__ normS,
                               float* __restrict__ agg) {
    int tid = blockIdx.x * blockDim.x + threadIdx.x;   // EE*NH threads
    int e = tid >> 6;
    int f = tid & 63;
    if (e < EE) {
        int s = src[e];
        int d = dst[e];
        float v = h[(size_t)s * NH + f] * normS[s];
        atomicAdd(&agg[(size_t)d * NH + f], v);
    }
}

// -- stage 4: x = relu(norm_dst*(agg@W) + b) -> f16, one node per 64-thr blk --
__global__ __launch_bounds__(64) void linear_relu_kernel(const float* __restrict__ agg,
                                                         const float* __restrict__ normD,
                                                         const float* __restrict__ W,
                                                         const float* __restrict__ bias,
                                                         _Float16* __restrict__ Xh) {
    __shared__ float a_sh[NH];
    const int n = blockIdx.x;
    const int j = threadIdx.x;
    a_sh[j] = agg[(size_t)n * NH + j];
    __syncthreads();
    float s = 0.0f;
#pragma unroll
    for (int k = 0; k < NH; ++k)
        s = fmaf(a_sh[k], W[k * NH + j], s);
    s = s * normD[n] + bias[j];
    s = fmaxf(s, 0.0f);
    Xh[(size_t)n * NH + j] = (_Float16)s;
}

// ---------------- stage 5: C = X * X^T via v_wmma_f32_16x16x32_f16 ----------
// Each wave computes a 32x32 output block (2x2 WMMA tiles, K=64 in 2 steps).
// Output is streamed with non-temporal stores (604 MB write-once >> 192 MB L2),
// keeping the 1.5 MB X matrix L2-resident for fragment loads.
__global__ __launch_bounds__(256) void xxt_wmma_kernel(const _Float16* __restrict__ X,
                                                       float* __restrict__ C) {
    const int lane = threadIdx.x & 31;
    const int wave = (blockIdx.x << 3) + (threadIdx.x >> 5);
    const int TPR  = NN / 32;               // 384 block-cols
    const int ty   = wave / TPR;
    const int tx   = wave % TPR;
    const int row0 = ty << 5;
    const int col0 = tx << 5;
    const int l    = lane & 15;
    const int hi   = lane >> 4;

    v8f c00 = {}, c01 = {}, c10 = {}, c11 = {};

    // A (16x32), contiguous k order chosen here: lanes 0-15 hold K 0..15, lanes 16-31 hold K 16..31
    // (matching B)  -> two 16B loads at +hi*16, +hi*16+8.
    const _Float16* arow0 = X + (size_t)(row0 + l) * NH + hi * 16;
    const _Float16* arow1 = arow0 + 16 * NH;
    // B (32x16), same contiguous k order as A: lanes 0-15 hold K 0..15, lanes 16-31 hold K 16..31 of
    // column N=l -> one contiguous 32B load from row (col0+l) of X.
    const _Float16* brow0 = X + (size_t)(col0 + l) * NH + hi * 16;
    const _Float16* brow1 = brow0 + 16 * NH;

#pragma unroll
    for (int kb = 0; kb < NH; kb += 32) {
        v8h a0lo = *(const v8h*)(arow0 + kb);
        v8h a0hi = *(const v8h*)(arow0 + kb + 8);
        v8h a1lo = *(const v8h*)(arow1 + kb);
        v8h a1hi = *(const v8h*)(arow1 + kb + 8);
        v16h a0 = __builtin_shufflevector(a0lo, a0hi, 0,1,2,3,4,5,6,7,8,9,10,11,12,13,14,15);
        v16h a1 = __builtin_shufflevector(a1lo, a1hi, 0,1,2,3,4,5,6,7,8,9,10,11,12,13,14,15);
        v16h b0 = *(const v16h*)(brow0 + kb);
        v16h b1 = *(const v16h*)(brow1 + kb);

        c00 = __builtin_amdgcn_wmma_f32_16x16x32_f16(false, a0, false, b0, (short)0, c00, false, false);
        c01 = __builtin_amdgcn_wmma_f32_16x16x32_f16(false, a0, false, b1, (short)0, c01, false, false);
        c10 = __builtin_amdgcn_wmma_f32_16x16x32_f16(false, a1, false, b0, (short)0, c10, false, false);
        c11 = __builtin_amdgcn_wmma_f32_16x16x32_f16(false, a1, false, b1, (short)0, c11, false, false);
    }

    // C/D f32 layout: VGPR r -> (M = r + 8*hi, N = l). Non-temporal streaming.
    float* out = C + (size_t)row0 * NN + col0;
#pragma unroll
    for (int r = 0; r < 8; ++r) {
        const int rr = r + (hi << 3);
        __builtin_nontemporal_store(c00[r], &out[(size_t)rr * NN + l]);
        __builtin_nontemporal_store(c01[r], &out[(size_t)rr * NN + 16 + l]);
        __builtin_nontemporal_store(c10[r], &out[(size_t)(rr + 16) * NN + l]);
        __builtin_nontemporal_store(c11[r], &out[(size_t)(rr + 16) * NN + 16 + l]);
    }
}

extern "C" void kernel_launch(void* const* d_in, const int* in_sizes, int n_in,
                              void* d_out, int out_size, void* d_ws, size_t ws_size,
                              hipStream_t stream) {
    (void)in_sizes; (void)n_in; (void)out_size; (void)ws_size;

    const float* h   = (const float*)d_in[0];   // [N, 64]
    const float* W   = (const float*)d_in[1];   // [64, 64]
    const float* b   = (const float*)d_in[2];   // [64]
    const int*   src = (const int*)d_in[3];     // [E]
    const int*   dst = (const int*)d_in[4];     // [E]
    float*       out = (float*)d_out;           // [N, N]

    float* ws   = (float*)d_ws;
    float* degO = ws;                           // N floats (becomes norm_src)
    float* degI = ws + NN;                      // N floats (becomes norm_dst)
    float* agg  = ws + 2 * NN;                  // N*64 floats
    _Float16* Xh = (_Float16*)(ws + 2 * NN + (size_t)NN * NH);  // N*64 halves, 32B aligned

    const int zn = 2 * NN + NN * NH;            // zero degrees + agg every call
    zero_kernel<<<(zn + 255) / 256, 256, 0, stream>>>(ws, zn);

    degree_kernel<<<(EE + 255) / 256, 256, 0, stream>>>(src, dst, degO, degI);
    norm_kernel<<<(NN + 255) / 256, 256, 0, stream>>>(degO, degI);

    scatter_kernel<<<(EE * NH) / 256, 256, 0, stream>>>(h, src, dst, degO, agg);

    linear_relu_kernel<<<NN, 64, 0, stream>>>(agg, degI, W, b, Xh);

    const int totalWaves = (NN / 32) * (NN / 32);   // 147456 waves, 8 per block
    xxt_wmma_kernel<<<totalWaves / 8, 256, 0, stream>>>(Xh, out);
}
